// DConvAttentionCorr_26542897889777
// MI455X (gfx1250) — matrix operand order repaired and checked by reading, not compiled, not run on hardware
//
#include <hip/hip_runtime.h>
#include <hip/hip_bf16.h>

// ---------------------------------------------------------------------------
// Types for WMMA fragments (gfx1250, wave32)
// ---------------------------------------------------------------------------
typedef __bf16 bf16x16 __attribute__((ext_vector_type(16)));
typedef float  f32x8   __attribute__((ext_vector_type(8)));
typedef float  f32x4   __attribute__((ext_vector_type(4)));
typedef unsigned int u32x4 __attribute__((ext_vector_type(4)));

union FragBF {
    bf16x16 v;
    u32x4   q[2];
};

#define ZERO8 {0.f,0.f,0.f,0.f,0.f,0.f,0.f,0.f}

// Problem constants
#define HH 128
#define WW 128
#define BB 4
#define M_TOT 65536            // B*H*W
#define K2 576                 // 64*9
#define N2 128
#define NOFF 32                // 18 off + 9 mask, padded to 32

#define WMMA_BF16(a, b, c) \
    __builtin_amdgcn_wmma_f32_16x16x32_bf16(false, (a), false, (b), (short)0, (c), false, false)

// ---------------------------------------------------------------------------
// Kernel 0: pack weights into B-fragment-ready bf16 tiles.
// B-frag layout used here (32x16 K-major tile; contiguous k, matches A, not ISA table order):
//   lane = (n%16) + 16*(ko/16), slot(bf16) = ko%16   (VGPR v holds K=2v,2v+1)
// Tile (kb,nb) stored as 32 lanes x 16 bf16 contiguous (1 KB).
// ---------------------------------------------------------------------------
__global__ void pack_weights(const float* __restrict__ w_off2, const float* __restrict__ b_off2,
                             const float* __restrict__ w_mask2, const float* __restrict__ b_mask2,
                             const float* __restrict__ w2,
                             __bf16* __restrict__ Bpk2, __bf16* __restrict__ Bpkoff,
                             float* __restrict__ boff_pad) {
    int t = blockIdx.x * blockDim.x + threadIdx.x;
    if (t < NOFF) {
        float bv = 0.f;
        if (t < 18) bv = b_off2[t];
        else if (t < 27) bv = b_mask2[t - 18];
        boff_pad[t] = bv;
    }
    if (t >= K2 * (N2 + NOFF)) return;
    int n = t / K2;
    int k = t - n * K2;
    int kb = k >> 5, ko = k & 31;
    int lane_hi = (ko >> 4) << 4;
    int slot = ko & 15;
    if (n < N2) {
        float v = w2[(size_t)n * K2 + k];
        int nb = n >> 4, no = n & 15;
        size_t d = ((size_t)(kb * 8 + nb) * 32 + (no + lane_hi)) * 16 + slot;
        Bpk2[d] = (__bf16)v;
    } else {
        int nn = n - N2;
        float v = 0.f;
        if (nn < 18)      v = w_off2[(size_t)nn * K2 + k];
        else if (nn < 27) v = w_mask2[(size_t)(nn - 18) * K2 + k];
        int nb = nn >> 4, no = nn & 15;
        size_t d = ((size_t)(kb * 2 + nb) * 32 + (no + lane_hi)) * 16 + slot;
        Bpkoff[d] = (__bf16)v;
    }
}

// ---------------------------------------------------------------------------
// Bilinear tap with zero padding
// ---------------------------------------------------------------------------
__device__ __forceinline__ float tap(const float* __restrict__ pl, int yy, int xx, float w) {
    if ((unsigned)yy < (unsigned)HH && (unsigned)xx < (unsigned)WW)
        return w * pl[(yy << 7) + xx];
    return 0.f;
}

// ---------------------------------------------------------------------------
// Kernel 1: layer-1 DCNv2 (C=3 -> 64), scalar f32 (K=27, negligible FLOPs).
// One thread per output pixel. h stored NCHW f32.
// ---------------------------------------------------------------------------
__global__ void layer1_kernel(const float* __restrict__ x,
                              const float* __restrict__ w_off1, const float* __restrict__ b_off1,
                              const float* __restrict__ w_mask1, const float* __restrict__ b_mask1,
                              const float* __restrict__ w1, const float* __restrict__ b1,
                              float* __restrict__ h) {
    const int m = blockIdx.x * 256 + threadIdx.x;
    const int b = m >> 14, y = (m >> 7) & 127, xq = m & 127;

    float xv[27];
#pragma unroll
    for (int c = 0; c < 3; ++c)
#pragma unroll
        for (int ky = 0; ky < 3; ++ky)
#pragma unroll
            for (int kx = 0; kx < 3; ++kx) {
                int yy = y + ky - 1, xx = xq + kx - 1;
                bool ok = ((unsigned)yy < (unsigned)HH) && ((unsigned)xx < (unsigned)WW);
                xv[c * 9 + ky * 3 + kx] = ok ? x[((((b * 3 + c) << 7) + yy) << 7) + xx] : 0.f;
            }

    float off[18];
    for (int j = 0; j < 18; ++j) {
        float a = b_off1[j];
#pragma unroll
        for (int i = 0; i < 27; ++i) a = fmaf(w_off1[j * 27 + i], xv[i], a);
        off[j] = a;
    }

    float samp[27];
    for (int kk = 0; kk < 9; ++kk) {
        float mkv = b_mask1[kk];
#pragma unroll
        for (int i = 0; i < 27; ++i) mkv = fmaf(w_mask1[kk * 27 + i], xv[i], mkv);
        float s = 1.f / (1.f + __expf(-mkv));
        int ky = kk / 3, kx = kk % 3;
        float py = off[2 * kk]     + (float)(y  - 1 + ky);
        float px = off[2 * kk + 1] + (float)(xq - 1 + kx);
        float fy = floorf(py), fx = floorf(px);
        int y0 = (int)fy, x0 = (int)fx;
        float wy1 = py - fy, wx1 = px - fx;
        float wy0 = 1.f - wy1, wx0 = 1.f - wx1;
        for (int c = 0; c < 3; ++c) {
            const float* pl = x + (((size_t)b * 3 + c) << 14);
            float acc = tap(pl, y0,     x0,     wy0 * wx0)
                      + tap(pl, y0,     x0 + 1, wy0 * wx1)
                      + tap(pl, y0 + 1, x0,     wy1 * wx0)
                      + tap(pl, y0 + 1, x0 + 1, wy1 * wx1);
            samp[c * 9 + kk] = acc * s;
        }
    }

    for (int oc = 0; oc < 64; ++oc) {
        float a = b1[oc];
#pragma unroll
        for (int i = 0; i < 27; ++i) a = fmaf(w1[oc * 27 + i], samp[i], a);
        h[(((size_t)b * 64 + oc) << 14) + (y << 7) + xq] = fmaxf(a, 0.f);
    }
}

// ---------------------------------------------------------------------------
// Kernel 2: regular im2col of h -> A (bf16 row-major [M][576], k = c*9+kk)
// grid (M/256, 64): blockIdx.y = channel (coalesced gathers along x)
// ---------------------------------------------------------------------------
__global__ void im2col_bf16(const float* __restrict__ h, __bf16* __restrict__ A) {
    int m = blockIdx.x * 256 + threadIdx.x;
    int c = blockIdx.y;
    int b = m >> 14, y = (m >> 7) & 127, xq = m & 127;
    const float* pl = h + (((size_t)b * 64 + c) << 14);
    __bf16* dst = A + (size_t)m * K2 + c * 9;
#pragma unroll
    for (int ky = 0; ky < 3; ++ky)
#pragma unroll
        for (int kx = 0; kx < 3; ++kx) {
            int yy = y + ky - 1, xx = xq + kx - 1;
            float v = ((unsigned)yy < (unsigned)HH && (unsigned)xx < (unsigned)WW)
                      ? pl[(yy << 7) + xx] : 0.f;
            dst[ky * 3 + kx] = (__bf16)v;
        }
}

// ---------------------------------------------------------------------------
// A-fragment load (16x32 A tile, contiguous k order used here, not ISA table order):
// lane L -> row L%16; lanes<16: K0..15 ; lanes>=16: K16..31 (matches B packing)
// From a row-major 64B row => two ds_load_b128 at +((L>=16)?32:0) and +16.
// ---------------------------------------------------------------------------
__device__ __forceinline__ void load_afrag(FragBF& f, const unsigned char* rowBase, int lane) {
    const unsigned char* p = rowBase + (((lane >> 4) & 1) << 5);
    f.q[0] = *(const u32x4*)p;
    f.q[1] = *(const u32x4*)(p + 16);
}

__device__ __forceinline__ void load_bfrag(FragBF& f, const __bf16* tile, int lane) {
    const u32x4* p = (const u32x4*)(tile + (size_t)lane * 16);
    f.q[0] = p[0];
    f.q[1] = p[1];
}

// ---------------------------------------------------------------------------
// Kernel 3: offset/mask conv for layer 2 as WMMA GEMM.
// M=65536, N=32 (18 off + 9 mask + pad), K=576.
// Block: 256 thr (8 waves), tile 128M x 32N; wave w: 16M x 32N.
// Double-buffered LDS A tile (80B row stride, bank-conflict-free) + register-
// pipelined B fragments: global loads for kb+1 are in flight under kb's WMMAs.
// ---------------------------------------------------------------------------
__global__ void gemm_off(const __bf16* __restrict__ A, const __bf16* __restrict__ Bpk,
                         const float* __restrict__ bias, float* __restrict__ Out27) {
    __shared__ __align__(16) unsigned char sA[2][128 * 80];
    const int tid = threadIdx.x;
    const int lane = tid & 31, wave = tid >> 5;
    const int m0 = blockIdx.x * 128;

    const int r0 = tid >> 2;            // rows 0..63
    const int r1 = r0 + 64;             // rows 64..127
    const int cB = (tid & 3) << 4;      // byte column within 64B row slice

    const unsigned char* A0 = (const unsigned char*)(A + (size_t)(m0 + r0) * K2);
    const unsigned char* A1 = (const unsigned char*)(A + (size_t)(m0 + r1) * K2);

    f32x8 acc[2] = { ZERO8, ZERO8 };
    const int aRow = lane & 15;

    // prologue: tile kb=0 -> LDS buf 0; B frags kb=0 in regs
    u32x4 st0 = *(const u32x4*)(A0 + cB);
    u32x4 st1 = *(const u32x4*)(A1 + cB);
    *(u32x4*)(sA[0] + r0 * 80 + cB) = st0;
    *(u32x4*)(sA[0] + r1 * 80 + cB) = st1;
    FragBF bf0, bf1;
    load_bfrag(bf0, Bpk + (size_t)0 * 512, lane);
    load_bfrag(bf1, Bpk + (size_t)1 * 512, lane);
    __syncthreads();

    for (int kb = 0; kb < 18; ++kb) {
        const int cur = kb & 1;
        const int nxt = (kb < 17) ? kb + 1 : kb;

        // issue next-tile global loads (complete under this step's WMMAs)
        st0 = *(const u32x4*)(A0 + nxt * 64 + cB);
        st1 = *(const u32x4*)(A1 + nxt * 64 + cB);
        FragBF bn0, bn1;
        load_bfrag(bn0, Bpk + (size_t)(nxt * 2 + 0) * 512, lane);
        load_bfrag(bn1, Bpk + (size_t)(nxt * 2 + 1) * 512, lane);
        if (kb < 16)
            __builtin_prefetch(A0 + (kb + 2) * 64 + cB, 0, 1);

        FragBF af;
        load_afrag(af, sA[cur] + (wave * 16 + aRow) * 80, lane);
        acc[0] = WMMA_BF16(af.v, bf0.v, acc[0]);
        acc[1] = WMMA_BF16(af.v, bf1.v, acc[1]);
        bf0 = bn0; bf1 = bn1;

        if (kb < 17) {
            *(u32x4*)(sA[cur ^ 1] + r0 * 80 + cB) = st0;
            *(u32x4*)(sA[cur ^ 1] + r1 * 80 + cB) = st1;
        }
        __syncthreads();
    }

    // store [m][32] row-major with bias (64B-coalesced per lane half)
    float bv0 = bias[lane & 15];
    float bv1 = bias[16 + (lane & 15)];
#pragma unroll
    for (int r = 0; r < 8; ++r) {
        int mrow = m0 + wave * 16 + ((lane >> 4) << 3) + r;
        Out27[(size_t)mrow * NOFF + (lane & 15)]      = acc[0][r] + bv0;
        Out27[(size_t)mrow * NOFF + 16 + (lane & 15)] = acc[1][r] + bv1;
    }
}

// ---------------------------------------------------------------------------
// Kernel 4: deformable sampler: sigmoid(mask) * bilinear(h), overwrite A (bf16)
// grid (M/256, 64); dy=row[2kk], dx=row[2kk+1], mask=row[18+kk].
// ---------------------------------------------------------------------------
__global__ void sampler_bf16(const float* __restrict__ h, const float* __restrict__ off27,
                             __bf16* __restrict__ A) {
    int m = blockIdx.x * 256 + threadIdx.x;
    int c = blockIdx.y;
    int b = m >> 14, y = (m >> 7) & 127, xq = m & 127;
    const float* r  = off27 + (size_t)m * NOFF;
    const float* pl = h + (((size_t)b * 64 + c) << 14);
    __bf16* dst = A + (size_t)m * K2 + c * 9;
#pragma unroll
    for (int kk = 0; kk < 9; ++kk) {
        float s  = 1.f / (1.f + __expf(-r[18 + kk]));
        float py = r[2 * kk]     + (float)(y  - 1 + kk / 3);
        float px = r[2 * kk + 1] + (float)(xq - 1 + kk % 3);
        float fy = floorf(py), fx = floorf(px);
        int y0 = (int)fy, x0 = (int)fx;
        float wy1 = py - fy, wx1 = px - fx;
        float wy0 = 1.f - wy1, wx0 = 1.f - wx1;
        float acc = tap(pl, y0,     x0,     wy0 * wx0)
                  + tap(pl, y0,     x0 + 1, wy0 * wx1)
                  + tap(pl, y0 + 1, x0,     wy1 * wx0)
                  + tap(pl, y0 + 1, x0 + 1, wy1 * wx1);
        dst[kk] = (__bf16)(acc * s);
    }
}

// ---------------------------------------------------------------------------
// Kernel 5: main GEMM: out = relu(A_def x W2 + b2), M=65536, N=128, K=576.
// Block 256 thr (8 waves), tile 128M x 128N (one full NCHW x-row per block).
// Wave tile 32M x 64N => 8 WMMA per K-step, 18 K-steps.
// Double-buffered LDS A + register-pipelined B.  Epilogue transposes through
// LDS (stride 133 floats = 5 mod 8 => conflict-free fragment writes), then
// contiguous float4 stores along x; bias+ReLU fused.
// ---------------------------------------------------------------------------
__global__ void gemm_main(const __bf16* __restrict__ A, const __bf16* __restrict__ Bpk,
                          const float* __restrict__ bias, float* __restrict__ out) {
    __shared__ __align__(16) unsigned char smem[128 * 133 * 4];  // 68 KB; A bufs use first 20 KB
    const int tid = threadIdx.x;
    const int lane = tid & 31, wave = tid >> 5;
    const int mw = wave >> 1, nw = wave & 1;
    const int m0 = blockIdx.x * 128;

    unsigned char* sA0 = smem;
    unsigned char* sA1 = smem + 128 * 80;

    const int r0 = tid >> 2;
    const int r1 = r0 + 64;
    const int cB = (tid & 3) << 4;

    const unsigned char* A0 = (const unsigned char*)(A + (size_t)(m0 + r0) * K2);
    const unsigned char* A1 = (const unsigned char*)(A + (size_t)(m0 + r1) * K2);

    f32x8 acc[2][4] = { { ZERO8, ZERO8, ZERO8, ZERO8 },
                        { ZERO8, ZERO8, ZERO8, ZERO8 } };
    const int aRow = lane & 15;

    // prologue
    u32x4 st0 = *(const u32x4*)(A0 + cB);
    u32x4 st1 = *(const u32x4*)(A1 + cB);
    *(u32x4*)(sA0 + r0 * 80 + cB) = st0;
    *(u32x4*)(sA0 + r1 * 80 + cB) = st1;
    FragBF bf[4];
#pragma unroll
    for (int j = 0; j < 4; ++j)
        load_bfrag(bf[j], Bpk + (size_t)(nw * 4 + j) * 512, lane);
    __syncthreads();

    for (int kb = 0; kb < 18; ++kb) {
        unsigned char* sCur = (kb & 1) ? sA1 : sA0;
        unsigned char* sNxt = (kb & 1) ? sA0 : sA1;
        const int nxt = (kb < 17) ? kb + 1 : kb;

        // issue next-tile global loads
        st0 = *(const u32x4*)(A0 + nxt * 64 + cB);
        st1 = *(const u32x4*)(A1 + nxt * 64 + cB);
        FragBF bn[4];
#pragma unroll
        for (int j = 0; j < 4; ++j)
            load_bfrag(bn[j], Bpk + (size_t)(nxt * 8 + nw * 4 + j) * 512, lane);
        if (kb < 16)
            __builtin_prefetch(A0 + (kb + 2) * 64 + cB, 0, 1);

        FragBF af[2];
#pragma unroll
        for (int f = 0; f < 2; ++f)
            load_afrag(af[f], sCur + (mw * 32 + f * 16 + aRow) * 80, lane);

#pragma unroll
        for (int j = 0; j < 4; ++j)
#pragma unroll
            for (int f = 0; f < 2; ++f)
                acc[f][j] = WMMA_BF16(af[f].v, bf[j].v, acc[f][j]);

#pragma unroll
        for (int j = 0; j < 4; ++j) bf[j] = bn[j];

        if (kb < 17) {
            *(u32x4*)(sNxt + r0 * 80 + cB) = st0;
            *(u32x4*)(sNxt + r1 * 80 + cB) = st1;
        }
        __syncthreads();
    }

    // epilogue: transpose through LDS, then contiguous stores along x
    float* Cs = (float*)smem;  // [n][m], stride 133
#pragma unroll
    for (int f = 0; f < 2; ++f)
#pragma unroll
        for (int j = 0; j < 4; ++j)
#pragma unroll
            for (int r = 0; r < 8; ++r) {
                int ml = mw * 32 + f * 16 + ((lane >> 4) << 3) + r;
                int nl = nw * 64 + j * 16 + (lane & 15);
                Cs[nl * 133 + ml] = acc[f][j][r];
            }
    __syncthreads();

    const int b = blockIdx.x >> 7, y = blockIdx.x & 127;
    for (int j = 0; j < 16; ++j) {
        int idx4 = tid + j * 256;
        int oc = idx4 >> 5;
        int x4 = (idx4 & 31) << 2;
        float bv = bias[oc];
        f32x4 v;
#pragma unroll
        for (int i = 0; i < 4; ++i)
            v[i] = fmaxf(Cs[oc * 133 + x4 + i] + bv, 0.f);
        *(f32x4*)(out + ((((size_t)b * 128 + oc) << 7) + y) * 128 + x4) = v;
    }
}

// ---------------------------------------------------------------------------
// Host launcher
// ---------------------------------------------------------------------------
extern "C" void kernel_launch(void* const* d_in, const int* in_sizes, int n_in,
                              void* d_out, int out_size, void* d_ws, size_t ws_size,
                              hipStream_t stream) {
    const float* x       = (const float*)d_in[0];
    const float* w_off1  = (const float*)d_in[1];
    const float* b_off1  = (const float*)d_in[2];
    const float* w_mask1 = (const float*)d_in[3];
    const float* b_mask1 = (const float*)d_in[4];
    const float* w1      = (const float*)d_in[5];
    const float* b1      = (const float*)d_in[6];
    const float* w_off2  = (const float*)d_in[7];
    const float* b_off2  = (const float*)d_in[8];
    const float* w_mask2 = (const float*)d_in[9];
    const float* b_mask2 = (const float*)d_in[10];
    const float* w2      = (const float*)d_in[11];
    const float* b2      = (const float*)d_in[12];
    float* out = (float*)d_out;

    unsigned char* ws = (unsigned char*)d_ws;
    float*  h       = (float*)(ws);                                  // 16,777,216 B
    __bf16* A       = (__bf16*)(ws + 16777216);                      // 75,497,472 B
    float*  Out27   = (float*)(ws + 16777216 + 75497472);            //  8,388,608 B
    __bf16* Bpk2    = (__bf16*)(ws + 100663296);                     //    147,456 B
    __bf16* Bpkoff  = (__bf16*)(ws + 100663296 + 147456);            //     36,864 B
    float*  boffpad = (float*)(ws + 100663296 + 147456 + 36864);     //        128 B

    pack_weights<<<(K2 * (N2 + NOFF) + 255) / 256, 256, 0, stream>>>(
        w_off2, b_off2, w_mask2, b_mask2, w2, Bpk2, Bpkoff, boffpad);

    layer1_kernel<<<M_TOT / 256, 256, 0, stream>>>(
        x, w_off1, b_off1, w_mask1, b_mask1, w1, b1, h);

    im2col_bf16<<<dim3(M_TOT / 256, 64), 256, 0, stream>>>(h, A);

    gemm_off<<<M_TOT / 128, 256, 0, stream>>>(A, Bpkoff, boffpad, Out27);

    sampler_bf16<<<dim3(M_TOT / 256, 64), 256, 0, stream>>>(h, Out27, A);

    gemm_main<<<M_TOT / 128, 256, 0, stream>>>(A, Bpk2, b2, out);
}
